// BUTDDecoder_16879221473802
// MI455X (gfx1250) — matrix operand order repaired and checked by reading, not compiled, not run on hardware
//
#include <hip/hip_runtime.h>
#include <hip/hip_bf16.h>

typedef __attribute__((ext_vector_type(16))) __bf16 v16bf;
typedef __attribute__((ext_vector_type(8)))  float  v8f;
typedef unsigned short u16;

#define DEV __device__ __forceinline__

constexpr int Bx = 128, Kobj = 36, VD = 2048, EMB = 1024, H = 1024, NT = 10000, ML = 20, TSTEPS = 19;
constexpr int H3 = 3 * H;

DEV u16 f2bf(float f) {
  unsigned int x = __float_as_uint(f);
  x += 0x7FFFu + ((x >> 16) & 1u);
  return (u16)(x >> 16);
}
DEV __bf16 bfbits(u16 u) { union { u16 s; __bf16 b; } c; c.s = u; return c.b; }
DEV float sigf(float x) { return 1.0f / (1.0f + __expf(-x)); }

// ---------------------------------------------------------------------------
// bf16 WMMA GEMM: C[M,N] = rowscale * act( A[M,K] * W[N,K]^T + addA + addB + bias )
// Each wave computes NTILES adjacent 16x16 output tiles (one shared A fragment,
// NTILES accumulators) -> NTILES x less A traffic from L2. Block = 4 waves.
// Fragment loads follow CDNA5 ISA 7.12.2 VGPR layouts (16-bit, 16x16x32).
// Inner K-loop is branchless (edge subtiles clamp their W pointer to row 0 and
// compute unconditionally; epilogue guard is scalar via readfirstlane), so
// EXEC stays all-ones for every WMMA and no exec save/restore in the hot loop.
// ---------------------------------------------------------------------------
template <int NTILES>
__global__ __launch_bounds__(128)
void gemm_bf16(const u16* __restrict__ A, int lda,
               const u16* __restrict__ W, int ldw,
               int M, int N, int K,
               const float* __restrict__ addA, int ldaddA,
               const float* __restrict__ addB, int ldaddB,
               const float* __restrict__ bias,
               const float* __restrict__ rowscale,
               int relu,
               float* __restrict__ Cf, long ldc,
               u16* __restrict__ Cbf, int ldcb)
{
  const int wave = __builtin_amdgcn_readfirstlane(threadIdx.x >> 5);
  const int lane = threadIdx.x & 31;
  const int m0 = blockIdx.y * 16;
  const int nb = __builtin_amdgcn_readfirstlane((blockIdx.x * 4 + wave) * (16 * NTILES));
  if (m0 >= M || nb >= N) return;            // scalar branch

  const int r  = lane & 15;
  const int hi = lane >> 4;
  const u16* Arow = A + (long)(m0 + r) * lda;

  const u16* Wrow[NTILES];
#pragma unroll
  for (int i = 0; i < NTILES; ++i) {
    // clamp out-of-range subtiles to row 0 (in-bounds); their results are never stored
    int nrow = (nb + 16 * i < N) ? (nb + 16 * i) : 0;
    Wrow[i] = W + (long)(nrow + r) * ldw;
  }

  v8f acc[NTILES];
#pragma unroll
  for (int i = 0; i < NTILES; ++i) acc[i] = v8f{};

  for (int k = 0; k < K; k += 32) {
    v16bf a;
#pragma unroll
    for (int j = 0; j < 8; ++j) {
      // A-matrix 16x32: M = lane&15; K = (j>>2)*16 + (lane>>4)*8 + (j&3)*2 + h
      int ka = k + ((j >> 2) << 4) + (hi << 3) + ((j & 3) << 1);
      a[2 * j]     = bfbits(Arow[ka]);
      a[2 * j + 1] = bfbits(Arow[ka + 1]);
    }
#pragma unroll
    for (int i = 0; i < NTILES; ++i) {
      v16bf b;
#pragma unroll
      for (int j = 0; j < 8; ++j) {
        // B-matrix 32x16: N = lane&15; K = (j>>2)*16 + (lane>>4)*8 + (j&3)*2 + h  (B = W^T)
        int kb = k + ((j >> 2) << 4) + (hi << 3) + ((j & 3) << 1);
        b[2 * j]     = bfbits(Wrow[i][kb]);
        b[2 * j + 1] = bfbits(Wrow[i][kb + 1]);
      }
      acc[i] = __builtin_amdgcn_wmma_f32_16x16x32_bf16(false, a, false, b,
                                                       (short)0, acc[i], false, false);
    }
  }

#pragma unroll
  for (int i = 0; i < NTILES; ++i) {
    if (nb + 16 * i >= N) break;             // scalar (nb uniform), taken at most once
    const int col = nb + 16 * i + r;
#pragma unroll
    for (int rr = 0; rr < 8; ++rr) {
      int row = m0 + rr + hi * 8;            // C layout: VGPR rr -> M = rr (+8 for lanes 16..31)
      float val = acc[i][rr];
      if (addA) val += addA[(long)row * ldaddA + col];
      if (addB) val += addB[(long)row * ldaddB + col];
      if (bias) val += bias[col];
      if (relu) val = fmaxf(val, 0.0f);
      if (rowscale) val *= rowscale[row];
      if (Cf)  Cf[(long)row * ldc + col] = val;
      if (Cbf) Cbf[(long)row * ldcb + col] = f2bf(val);
    }
  }
}

// --------------------------- small helper kernels ---------------------------

__global__ void zero_u32(unsigned int* p, long n) {
  long i = (long)blockIdx.x * blockDim.x + threadIdx.x;
  if (i < n) p[i] = 0u;
}

__global__ void convert_bf16(const float* __restrict__ s, u16* __restrict__ d, long n) {
  long i = (long)blockIdx.x * blockDim.x + threadIdx.x;
  if (i < n) d[i] = f2bf(s[i]);
}

// dst[b*rowsz + j] = bf16(src[order[b]*rowsz + j])
__global__ void gather_bf16(const float* __restrict__ s, const int* __restrict__ order,
                            u16* __restrict__ d, long rowsz, int nrows) {
  long i = (long)blockIdx.x * blockDim.x + threadIdx.x;
  if (i >= (long)nrows * rowsz) return;
  long b = i / rowsz;
  long j = i - b * rowsz;
  d[i] = f2bf(s[(long)order[b] * rowsz + j]);
}

// stable descending argsort of cap_len (B=128), sorted lengths + time masks
__global__ void order_kernel(const int* __restrict__ cap_len, int* __restrict__ order,
                             int* __restrict__ cap_sorted, float* __restrict__ maskbuf) {
  __shared__ int s_order[Bx];
  int i = threadIdx.x;
  int li = cap_len[i];
  int rank = 0;
  for (int j = 0; j < Bx; ++j) {
    int lj = cap_len[j];
    if (lj > li || (lj == li && j < i)) rank++;
  }
  s_order[rank] = i;
  __syncthreads();
  int src = s_order[i];
  order[i] = src;
  int cl = cap_len[src];
  cap_sorted[i] = cl;
  for (int t = 0; t < TSTEPS; ++t)
    maskbuf[t * Bx + i] = (t < cl - 1) ? 1.0f : 0.0f;
}

__global__ void vmean_kernel(const float* __restrict__ v, const int* __restrict__ order,
                             u16* __restrict__ vm) {
  long i = (long)blockIdx.x * blockDim.x + threadIdx.x;
  if (i >= (long)Bx * VD) return;
  int b = (int)(i >> 11);
  int d = (int)(i & (VD - 1));
  const float* vb = v + (long)order[b] * Kobj * VD + d;
  float s = 0.0f;
  for (int k = 0; k < Kobj; ++k) s += vb[(long)k * VD];
  vm[i] = f2bf(s * (1.0f / (float)Kobj));
}

// PyTorch GRUCell combine: gi/gh already include biases
__global__ void gru_combine(const float* __restrict__ gi, const float* __restrict__ gh,
                            float* __restrict__ h, u16* __restrict__ h_bf) {
  long i = (long)blockIdx.x * blockDim.x + threadIdx.x;
  if (i >= (long)Bx * H) return;
  int b = (int)(i / H);
  int j = (int)(i - (long)b * H);
  long base = (long)b * H3 + j;
  float r  = sigf(gi[base]           + gh[base]);
  float z  = sigf(gi[base + H]       + gh[base + H]);
  float nn = tanhf(gi[base + 2 * H]  + r * gh[base + 2 * H]);
  float hv = h[i];
  float out = (1.0f - z) * nn + z * hv;
  h[i] = out;
  h_bf[i] = f2bf(out);
}

// logits[b,k] = sum_h vproj[b,k,h] * q[b,h] * wa[h] + ba   (one wave per (b,k) row)
__global__ void attn_logits(const float* __restrict__ vproj, const float* __restrict__ q,
                            const float* __restrict__ wa, const float* __restrict__ ba,
                            float* __restrict__ logits) {
  int row = blockIdx.x * 4 + (threadIdx.x >> 5);
  if (row >= Bx * Kobj) return;
  int lane = threadIdx.x & 31;
  int b = row / Kobj;
  const float* vp = vproj + (long)row * H;
  const float* qb = q + (long)b * H;
  float acc = 0.0f;
  for (int hh = lane; hh < H; hh += 32) acc += vp[hh] * qb[hh] * wa[hh];
  for (int off = 16; off; off >>= 1) acc += __shfl_down(acc, off, 32);
  if (lane == 0) logits[row] = acc + ba[0];
}

// softmax over K=36, write masked alphas, and attended feature att_v (bf16)
__global__ void attn_sm_av(const float* __restrict__ logits, const float* __restrict__ v,
                           const int* __restrict__ order, const float* __restrict__ maskrow,
                           float* __restrict__ alphas, u16* __restrict__ attv_bf) {
  __shared__ float att[Kobj];
  int b = blockIdx.x;
  float m = maskrow[b];
  if (threadIdx.x == 0) {
    float mx = -1e30f;
    for (int k = 0; k < Kobj; ++k) mx = fmaxf(mx, logits[b * Kobj + k]);
    float s = 0.0f;
    for (int k = 0; k < Kobj; ++k) { float e = __expf(logits[b * Kobj + k] - mx); att[k] = e; s += e; }
    float inv = 1.0f / s;
    for (int k = 0; k < Kobj; ++k) att[k] *= inv;
  }
  __syncthreads();
  if (threadIdx.x < Kobj)
    alphas[(long)b * (ML * Kobj) + threadIdx.x] = att[threadIdx.x] * m;
  const float* vb = v + (long)order[b] * Kobj * VD;
  for (int d = threadIdx.x; d < VD; d += blockDim.x) {
    float s = 0.0f;
    for (int k = 0; k < Kobj; ++k) s += att[k] * vb[(long)k * VD + d];
    attv_bf[(long)b * VD + d] = f2bf(s);
  }
}

// ------------------------------- host driver --------------------------------

static void launch_gemm(hipStream_t st, const u16* A, int lda, const u16* W, int ldw,
                        int M, int N, int K, const float* addA, int ldaddA,
                        const float* addB, int ldaddB, const float* bias,
                        const float* rowscale, int relu,
                        float* Cf, long ldc, u16* Cbf, int ldcb) {
  if (N >= 2048) {
    // 4 tiles per wave: block covers 256 columns; less redundant A traffic
    dim3 g((N + 255) / 256, (M + 15) / 16);
    gemm_bf16<4><<<g, 128, 0, st>>>(A, lda, W, ldw, M, N, K, addA, ldaddA, addB, ldaddB,
                                    bias, rowscale, relu, Cf, ldc, Cbf, ldcb);
  } else {
    // 1 tile per wave: maximum block-level parallelism for small latency-bound GEMMs
    dim3 g((N + 63) / 64, (M + 15) / 16);
    gemm_bf16<1><<<g, 128, 0, st>>>(A, lda, W, ldw, M, N, K, addA, ldaddA, addB, ldaddB,
                                    bias, rowscale, relu, Cf, ldc, Cbf, ldcb);
  }
}

static inline int gsz(long n, int b) { return (int)((n + b - 1) / b); }

extern "C" void kernel_launch(void* const* d_in, const int* in_sizes, int n_in,
                              void* d_out, int out_size, void* d_ws, size_t ws_size,
                              hipStream_t stream) {
  const float* v       = (const float*)d_in[0];
  const float* caption = (const float*)d_in[1];
  const int*   cap_len = (const int*)d_in[2];
  const float* Wih1 = (const float*)d_in[3];
  const float* Whh1 = (const float*)d_in[4];
  const float* bih1 = (const float*)d_in[5];
  const float* bhh1 = (const float*)d_in[6];
  const float* Wih2 = (const float*)d_in[7];
  const float* Whh2 = (const float*)d_in[8];
  const float* bih2 = (const float*)d_in[9];
  const float* bhh2 = (const float*)d_in[10];
  const float* Wv = (const float*)d_in[11];
  const float* bv = (const float*)d_in[12];
  const float* Wq = (const float*)d_in[13];
  const float* bq = (const float*)d_in[14];
  const float* wa = (const float*)d_in[15];
  const float* ba = (const float*)d_in[16];
  const float* W1 = (const float*)d_in[17];
  const float* b1 = (const float*)d_in[18];
  const float* W2 = (const float*)d_in[19];
  const float* b2 = (const float*)d_in[20];

  float* out = (float*)d_out;
  float* predict = out;                                   // [B, ML, NT]
  float* alphas  = out + (long)Bx * ML * NT;              // [B, ML, K]

  // ---- workspace carve-up ----
  char* w = (char*)d_ws;
  size_t off = 0;
  auto alloc = [&](size_t bytes) -> void* {
    void* p = w + off;
    off = (off + bytes + 255) & ~(size_t)255;
    return p;
  };
  u16* wih1_bf = (u16*)alloc((size_t)H3 * 4096 * 2);
  u16* whh1_bf = (u16*)alloc((size_t)H3 * H * 2);
  u16* wih2_bf = (u16*)alloc((size_t)H3 * 3072 * 2);
  u16* whh2_bf = (u16*)alloc((size_t)H3 * H * 2);
  u16* wv_bf   = (u16*)alloc((size_t)H * VD * 2);
  u16* wq_bf   = (u16*)alloc((size_t)H * H * 2);
  u16* w1_bf   = (u16*)alloc((size_t)H * H * 2);
  u16* w2_bf   = (u16*)alloc((size_t)NT * H * 2);
  u16* v_bf    = (u16*)alloc((size_t)Bx * Kobj * VD * 2); // sorted
  u16* cap_bf  = (u16*)alloc((size_t)Bx * ML * EMB * 2);  // sorted
  u16* vmean_bf= (u16*)alloc((size_t)Bx * VD * 2);
  float* pre1  = (float*)alloc((size_t)Bx * H3 * 4);      // v_mean part + bih1
  float* Pbuf  = (float*)alloc((size_t)Bx * ML * H3 * 4); // prev-embedding parts
  float* vproj = (float*)alloc((size_t)Bx * Kobj * H * 4);
  float* gi1   = (float*)alloc((size_t)Bx * H3 * 4);
  float* gh1   = (float*)alloc((size_t)Bx * H3 * 4);
  float* gi2   = (float*)alloc((size_t)Bx * H3 * 4);
  float* gh2   = (float*)alloc((size_t)Bx * H3 * 4);
  float* h1f   = (float*)alloc((size_t)Bx * H * 4);
  float* h2f   = (float*)alloc((size_t)Bx * H * 4);
  u16*  h1bf   = (u16*)alloc((size_t)Bx * H * 2);
  u16*  h2bf   = (u16*)alloc((size_t)Bx * H * 2);
  float* hqf   = (float*)alloc((size_t)Bx * H * 4);
  u16*  hq_bf  = (u16*)alloc((size_t)Bx * H * 2);
  float* qf    = (float*)alloc((size_t)Bx * H * 4);
  u16*  attv_bf= (u16*)alloc((size_t)Bx * VD * 2);
  float* logit = (float*)alloc((size_t)Bx * Kobj * 4);
  int*  order  = (int*)alloc(Bx * 4);
  int*  cap_s  = (int*)alloc(Bx * 4);
  float* maskb = (float*)alloc((size_t)TSTEPS * Bx * 4);
  (void)ws_size; (void)n_in; (void)in_sizes;

  // ---- init ----
  long out_words = (long)Bx * ML * NT + (long)Bx * ML * Kobj;
  zero_u32<<<gsz(out_words, 256), 256, 0, stream>>>((unsigned int*)out, out_words);
  zero_u32<<<gsz(Bx * H, 256), 256, 0, stream>>>((unsigned int*)h1f, Bx * H);
  zero_u32<<<gsz(Bx * H, 256), 256, 0, stream>>>((unsigned int*)h2f, Bx * H);
  zero_u32<<<gsz(Bx * H / 2, 256), 256, 0, stream>>>((unsigned int*)h1bf, Bx * H / 2);
  zero_u32<<<gsz(Bx * H / 2, 256), 256, 0, stream>>>((unsigned int*)h2bf, Bx * H / 2);

  order_kernel<<<1, Bx, 0, stream>>>(cap_len, order, cap_s, maskb);

  // ---- stage weights in bf16 (stay resident in 192MB L2 across the recurrence) ----
  auto cvt = [&](const float* s, u16* d, long n) {
    convert_bf16<<<gsz(n, 256), 256, 0, stream>>>(s, d, n);
  };
  cvt(Wih1, wih1_bf, (long)H3 * 4096);
  cvt(Whh1, whh1_bf, (long)H3 * H);
  cvt(Wih2, wih2_bf, (long)H3 * 3072);
  cvt(Whh2, whh2_bf, (long)H3 * H);
  cvt(Wv,   wv_bf,   (long)H * VD);
  cvt(Wq,   wq_bf,   (long)H * H);
  cvt(W1,   w1_bf,   (long)H * H);
  cvt(W2,   w2_bf,   (long)NT * H);

  gather_bf16<<<gsz((long)Bx * Kobj * VD, 256), 256, 0, stream>>>(v, order, v_bf, (long)Kobj * VD, Bx);
  gather_bf16<<<gsz((long)Bx * ML * EMB, 256), 256, 0, stream>>>(caption, order, cap_bf, (long)ML * EMB, Bx);
  vmean_kernel<<<gsz((long)Bx * VD, 256), 256, 0, stream>>>(v, order, vmean_bf);

  // ---- hoisted GEMMs ----
  // vproj = relu(v_s @ Wv^T + bv)                       [B*K, H]
  launch_gemm(stream, v_bf, VD, wv_bf, VD, Bx * Kobj, H, VD,
              nullptr, 0, nullptr, 0, bv, nullptr, 1, vproj, H, nullptr, 0);
  // pre1 = v_mean @ Wih1[:,H:H+VD]^T + bih1            [B, 3H]
  launch_gemm(stream, vmean_bf, VD, wih1_bf + H, 4096, Bx, H3, VD,
              nullptr, 0, nullptr, 0, bih1, nullptr, 0, pre1, H3, nullptr, 0);
  // Pbuf[b,t] = caption_s[b,t] @ Wih1[:,H+VD:]^T       [B*ML, 3H]
  launch_gemm(stream, cap_bf, EMB, wih1_bf + (H + VD), 4096, Bx * ML, H3, EMB,
              nullptr, 0, nullptr, 0, nullptr, nullptr, 0, Pbuf, H3, nullptr, 0);

  // ---- sequential decode ----
  for (int t = 0; t < TSTEPS; ++t) {
    const float* mrow = maskb + (long)t * Bx;
    // gi1 = h2 @ Wih1[:, :H]^T + pre1 + Pbuf[:, t]
    launch_gemm(stream, h2bf, H, wih1_bf, 4096, Bx, H3, H,
                pre1, H3, Pbuf + (long)t * H3, ML * H3, nullptr, nullptr, 0, gi1, H3, nullptr, 0);
    // gh1 = h1 @ Whh1^T + bhh1
    launch_gemm(stream, h1bf, H, whh1_bf, H, Bx, H3, H,
                nullptr, 0, nullptr, 0, bhh1, nullptr, 0, gh1, H3, nullptr, 0);
    gru_combine<<<gsz((long)Bx * H, 256), 256, 0, stream>>>(gi1, gh1, h1f, h1bf);
    // hq = h1n @ W1^T + b1
    launch_gemm(stream, h1bf, H, w1_bf, H, Bx, H, H,
                nullptr, 0, nullptr, 0, b1, nullptr, 0, hqf, H, hq_bf, H);
    // q = relu(hq @ Wq^T + bq)
    launch_gemm(stream, hq_bf, H, wq_bf, H, Bx, H, H,
                nullptr, 0, nullptr, 0, bq, nullptr, 1, qf, H, nullptr, 0);
    attn_logits<<<gsz(Bx * Kobj, 4), 128, 0, stream>>>(vproj, qf, wa, ba, logit);
    attn_sm_av<<<Bx, 256, 0, stream>>>(logit, v, order, mrow,
                                       alphas + (long)t * Kobj, attv_bf);
    // gi2 = att_v @ Wih2[:, :VD]^T + bih2 ; += hq @ Wih2[:, VD:]^T
    launch_gemm(stream, attv_bf, VD, wih2_bf, 3072, Bx, H3, VD,
                nullptr, 0, nullptr, 0, bih2, nullptr, 0, gi2, H3, nullptr, 0);
    launch_gemm(stream, hq_bf, H, wih2_bf + VD, 3072, Bx, H3, H,
                gi2, H3, nullptr, 0, nullptr, nullptr, 0, gi2, H3, nullptr, 0);
    // gh2 = h2 @ Whh2^T + bhh2
    launch_gemm(stream, h2bf, H, whh2_bf, H, Bx, H3, H,
                nullptr, 0, nullptr, 0, bhh2, nullptr, 0, gh2, H3, nullptr, 0);
    gru_combine<<<gsz((long)Bx * H, 256), 256, 0, stream>>>(gi2, gh2, h2f, h2bf);
    // predict[:, t, :] = mask * (h2n @ W2^T + b2)
    launch_gemm(stream, h2bf, H, w2_bf, H, Bx, NT, H,
                nullptr, 0, nullptr, 0, b2, mrow, 0,
                predict + (long)t * NT, (long)ML * NT, nullptr, 0);
  }
}
